// ChannelDropout_5626407158062
// MI455X (gfx1250) — compile-verified, matrix operand order checked by reading, not run on hardware
//
#include <hip/hip_runtime.h>

typedef __attribute__((ext_vector_type(16))) _Float16 v16h;
typedef __attribute__((ext_vector_type(8)))  float    v8f;
typedef __attribute__((ext_vector_type(4)))  float    f4;

#define DROPOUT2 0.04f   // 0.2^2, compare squared distances (avoids sqrt)
#define EPS      1e-8f

// ---------------------------------------------------------------------------
// Phase 1: per-channel scale = kept(center) / (EPS + mean_k kept(mc[k]))
// One wave32 per tile of 16 channels. The 100-trial reduction is done with
// V_WMMA_F32_16X16X32_F16: A[m,k] = MC keep flag (exact 0/1 in f16),
// B = ones, accumulated over 4 K-blocks (K=0..127, k>=NMC zeroed).
// ---------------------------------------------------------------------------
__global__ __launch_bounds__(32) void ChannelDropout_scale_kernel(
    const float* __restrict__ positions,  // (BC, 2)
    const float* __restrict__ center,     // (2,)
    const float* __restrict__ mc,         // (NMC, 2)
    float* __restrict__ scale,            // (BC,)
    int BC, int NMC)
{
    __shared__ float sums[16];

    const int  lane = threadIdx.x;        // 0..31, full wave
    const int  m    = lane & 15;          // channel row within tile
    const bool hi   = lane >= 16;         // high half-wave owns K+8 / K+16 slots
    const int  ch   = blockIdx.x * 16 + m;
    const int  chc  = (ch < BC) ? ch : (BC - 1);   // clamped (keeps EXEC full)

    const float px = positions[chc * 2 + 0];
    const float py = positions[chc * 2 + 1];

    // B matrix: all ones (any layout of all-ones is the ones matrix)
    v16h b;
#pragma unroll
    for (int j = 0; j < 16; ++j) b[j] = (_Float16)1.0f;

    v8f c = {};
#pragma unroll
    for (int kb = 0; kb < 128; kb += 32) {   // 4 K-blocks cover NMC=100
        v16h a;
#pragma unroll
        for (int j = 0; j < 16; ++j) {
            // ISA 16-bit A 16x32 layout:
            //  low lanes : VGPR0-3 -> K=0..7,  VGPR4-7 -> K=16..23
            //  high lanes: VGPR0-3 -> K=8..15, VGPR4-7 -> K=24..31
            const int kl = (j < 8 ? j : j + 8) + (hi ? 8 : 0);
            const int k  = kb + kl;
            const int kc = (k < NMC) ? k : 0;
            const float dx = px - mc[kc * 2 + 0];
            const float dy = py - mc[kc * 2 + 1];
            const bool keep = (k < NMC) && (dx * dx + dy * dy > DROPOUT2);
            a[j] = keep ? (_Float16)1.0f : (_Float16)0.0f;
        }
        // EXEC is all-ones here (no divergence above, only selects)
        c = __builtin_amdgcn_wmma_f32_16x16x32_f16(
                /*neg_a=*/false, a, /*neg_b=*/false, b,
                /*c_mod=*/(short)0, c, /*reuse_a=*/false, /*reuse_b=*/false);
    }

    // D layout: lane 0 holds D[M=r, N=0] in c[r] (channels 0..7),
    //           lane 16 holds D[M=8+r, N=0] in c[r] (channels 8..15).
    if (lane == 0) {
#pragma unroll
        for (int r = 0; r < 8; ++r) sums[r] = c[r];
    }
    if (lane == 16) {
#pragma unroll
        for (int r = 0; r < 8; ++r) sums[8 + r] = c[r];
    }
    __syncthreads();

    if (lane < 16 && ch < BC) {
        const float proba = sums[lane] / (float)NMC;
        const float dx = px - center[0];
        const float dy = py - center[1];
        const bool kept = (dx * dx + dy * dy > DROPOUT2);
        scale[ch] = kept ? (1.0f / (EPS + proba)) : 0.0f;
    }
}

// ---------------------------------------------------------------------------
// Phase 2: out[ch, t] = sig[ch, t] * scale[ch]. Pure HBM stream (~419 MB
// total -> ~18 us at 23.3 TB/s). float4 (b128) accesses, non-temporal hints
// (read-once/write-once; working set > 192 MB L2), scalar scale per block.
// ---------------------------------------------------------------------------
__global__ __launch_bounds__(256) void ChannelDropout_apply_kernel(
    const f4* __restrict__ sig,
    const float* __restrict__ scale,
    f4* __restrict__ out,
    int T4)
{
    const int chan = blockIdx.y;
    const int idx  = blockIdx.x * 256 + threadIdx.x;
    if (idx < T4) {
        const float  s   = scale[chan];       // block-uniform -> s_load
        const size_t off = (size_t)chan * (size_t)T4 + (size_t)idx;
        f4 v = __builtin_nontemporal_load(&sig[off]);
        v *= s;
        __builtin_nontemporal_store(v, &out[off]);
    }
}

extern "C" void kernel_launch(void* const* d_in, const int* in_sizes, int n_in,
                              void* d_out, int out_size, void* d_ws, size_t ws_size,
                              hipStream_t stream) {
    const float* sig       = (const float*)d_in[0];   // (B, C, T) f32
    const float* positions = (const float*)d_in[1];   // (B, C, 2) f32
    const float* center    = (const float*)d_in[2];   // (2,)      f32
    const float* mc        = (const float*)d_in[3];   // (N, 2)    f32
    float*       out       = (float*)d_out;
    float*       scale     = (float*)d_ws;            // BC floats of scratch

    const int BC  = in_sizes[1] / 2;     // 64*273 = 17472
    const int T   = in_sizes[0] / BC;    // 3000
    const int NMC = in_sizes[3] / 2;     // 100
    const int T4  = T / 4;               // 750 (T divisible by 4)

    // Phase 1: 16 channels per wave32 block, WMMA reduction over MC trials
    const int tiles = (BC + 15) / 16;    // 1092
    ChannelDropout_scale_kernel<<<tiles, 32, 0, stream>>>(
        positions, center, mc, scale, BC, NMC);

    // Phase 2: bandwidth-bound scaling stream
    dim3 grid((T4 + 255) / 256, BC);     // (3, 17472)
    ChannelDropout_apply_kernel<<<grid, 256, 0, stream>>>(
        (const f4*)sig, scale, (f4*)out, T4);
}
